// CubicRNN_43490838839736
// MI455X (gfx1250) — compile-verified, matrix operand order checked by reading, not run on hardware
//
#include <hip/hip_runtime.h>

typedef __attribute__((ext_vector_type(16))) _Float16 v16h;
typedef __attribute__((ext_vector_type(8)))  _Float16 v8h;
typedef __attribute__((ext_vector_type(8)))  float    v8f;

// Problem constants (from setup_inputs): B=4,T=10,C=1,H=W=64,HC=32,S=3,L=3
#define NPIX   16384        // B*H*W
#define CINP0  96           // padded cat channels (layers 0..2)
#define CINY   64           // conv_y input channels
#define NGATE  128          // 4*HC

union ABu { v16h v; v8h h[2]; };

// ---------------------------------------------------------------------------
// Implicit-GEMM conv via WMMA.
//   A = activations (M = pixels x K = k*k*CIN, NHWC f16, SAME padding via
//       predicated per-lane global_load_b128 gathers).
//   B = pre-packed weight fragments, staged per 32-wide K-chunk through LDS
//       (double-buffered, register-staged pipeline; each chunk loaded once
//       per block and shared by all 8 waves).
// Block = 256 thr = 8 waves; M-tile = 128 pixels (2 image rows); each wave
// owns one 16-pixel M-subtile and computes all COUT/16 N-tiles.
// ---------------------------------------------------------------------------
template<int KW, int CIN, int COUT, bool OUT_F16>
__global__ __launch_bounds__(256)
void conv_wmma(const _Float16* __restrict__ act,
               const _Float16* __restrict__ wpk,
               const float* __restrict__ bias,
               void* __restrict__ out)
{
  constexpr int CPS = CIN / 32;          // K-chunks per spatial tap
  constexpr int KC  = KW * KW * CPS;     // total K-chunks of 32
  constexpr int NBT = COUT / 16;         // N blocks of 16
  constexpr int PAD = KW / 2;
  constexpr int CHUNK = NBT * 512;       // f16 elements per staged chunk
  constexpr int PT  = CHUNK / 256;       // f16 copied per thread per chunk

  __shared__ __align__(32) _Float16 lds[2 * CHUNK];

  const int lane = threadIdx.x & 31;
  const int wave = threadIdx.x >> 5;
  const int r    = blockIdx.x;           // 0..127 : 2 rows each
  const int row0 = r << 1;
  const int b    = row0 >> 6;
  const int y0   = row0 & 63;
  const int row  = y0 + (wave >> 2);     // this wave's image row
  const int xb   = (wave & 3) << 4;      // this wave's 16-pixel column block
  const int m    = lane & 15;
  const int hlf  = lane >> 4;
  const int x    = xb + m;
  const int cb   = hlf * 8;

  v8f zf = {};
  v8f acc[NBT];
#pragma unroll
  for (int t = 0; t < NBT; ++t) acc[t] = zf;

  // ---- prologue: stage chunk 0 into LDS buffer 0 ----
  _Float16 regs[PT];
  {
    const _Float16* src = wpk + (size_t)threadIdx.x * PT;
#pragma unroll
    for (int i = 0; i < PT; ++i) regs[i] = src[i];
#pragma unroll
    for (int i = 0; i < PT; ++i) lds[threadIdx.x * PT + i] = regs[i];
  }
  __syncthreads();

  int buf = 0;
  for (int kc = 0; kc < KC; ++kc) {
    // prefetch next chunk into registers (overlaps WMMA below)
    if (kc + 1 < KC) {
      const _Float16* src = wpk + (size_t)(kc + 1) * CHUNK + (size_t)threadIdx.x * PT;
#pragma unroll
      for (int i = 0; i < PT; ++i) regs[i] = src[i];
    }

    // A fragment: implicit-GEMM gather for this K-chunk
    const int sp = kc / CPS;
    const int cc = (kc - sp * CPS) * 32 + cb;
    const int dy = sp / KW, dx = sp - dy * KW;
    const int yy = row + dy - PAD, xx = x + dx - PAD;
    ABu a;
    if ((unsigned)yy < 64u && (unsigned)xx < 64u) {
      const _Float16* p = act + (size_t)((((b << 6) + yy) << 6) + xx) * CIN + cc;
      a.h[0] = *(const v8h*)(p);
      a.h[1] = *(const v8h*)(p + 16);
    } else {
      v16h z = {};
      a.v = z;
    }

    // compute: all N-tiles from LDS-resident B fragments
    const _Float16* lbase = lds + buf * CHUNK + lane * 16;
#pragma unroll
    for (int t = 0; t < NBT; ++t) {
      const v16h bf = *(const v16h*)(lbase + t * 512);
      acc[t] = __builtin_amdgcn_wmma_f32_16x16x32_f16(
          false, a.v, false, bf, (short)0, acc[t], false, false);
    }

    // write prefetched chunk into the other buffer; one barrier per chunk
    if (kc + 1 < KC) {
      _Float16* dst = lds + (buf ^ 1) * CHUNK + threadIdx.x * PT;
#pragma unroll
      for (int i = 0; i < PT; ++i) dst[i] = regs[i];
      buf ^= 1;
    }
    __syncthreads();
  }

  // Epilogue. C-layout: vgpr rr, lane l -> M = rr + (l>>4)*8, N = l&15.
  const int pix0 = (((b << 6) + row) << 6) + xb;
#pragma unroll
  for (int t = 0; t < NBT; ++t) {
    const int n = t * 16 + (lane & 15);
    const float bs = bias ? bias[n] : 0.f;
#pragma unroll
    for (int rr = 0; rr < 8; ++rr) {
      const int mm = rr + hlf * 8;
      const size_t o = (size_t)(pix0 + mm) * COUT + n;
      const float v = acc[t][rr] + bs;
      if constexpr (OUT_F16) ((_Float16*)out)[o] = (_Float16)v;
      else                   ((float*)out)[o] = v;
    }
  }
}

// ---------------------------------------------------------------------------
// Weight packer: OIHW f32 -> per-lane WMMA B-fragment order, K = (dy,dx,c),
// channels zero-padded CINSRC -> CINP. Fragment elem e of lane:
//   K = kc*32 + (lane>>4)*8 + (e<8 ? e : e+8)
// ---------------------------------------------------------------------------
__global__ void pack_weights(const float* __restrict__ w, _Float16* __restrict__ dst,
                             int COUT, int CINSRC, int KW, int CINP, int total)
{
  int i = blockIdx.x * blockDim.x + threadIdx.x;
  if (i >= total) return;
  const int NBT = COUT >> 4;
  const int e    = i & 15;
  const int lane = (i >> 4) & 31;
  const int nb   = (i >> 9) % NBT;
  const int kc   = i / (NBT * 512);
  const int n    = nb * 16 + (lane & 15);
  const int k    = kc * 32 + (lane >> 4) * 8 + (e < 8 ? e : e + 8);
  const int sp   = k / CINP;
  const int c    = k - sp * CINP;
  const int dy   = sp / KW, dx = sp - dy * KW;
  float v = 0.f;
  if (c < CINSRC)
    v = w[(((size_t)n * CINSRC + c) * KW + dy) * KW + dx];
  dst[i] = (_Float16)v;
}

// ---------------------------------------------------------------------------
// Elementwise LSTM gating for both branches; writes new states and ycat.
// ---------------------------------------------------------------------------
__device__ __forceinline__ float sigm(float v) { return 1.f / (1.f + __expf(-v)); }
__device__ __forceinline__ float tanhx(float v) { return 1.f - 2.f / (__expf(2.f * v) + 1.f); }

__global__ void gate_kernel(const float* __restrict__ gx, const float* __restrict__ gz,
                            const float* __restrict__ cxin, float* __restrict__ cxout,
                            float* __restrict__ czio,
                            _Float16* __restrict__ hxout, _Float16* __restrict__ hzio,
                            _Float16* __restrict__ ycat)
{
  int i = blockIdx.x * blockDim.x + threadIdx.x;
  if (i >= NPIX * 32) return;
  const int p = i >> 5, c = i & 31;
  const float* g = gx + (size_t)p * NGATE;
  float cn = sigm(g[32 + c] + 0.01f) * cxin[i] + sigm(g[c]) * tanhx(g[64 + c]);
  float h  = sigm(g[96 + c]) * tanhx(cn);
  cxout[i] = cn;
  hxout[i] = (_Float16)h;
  ycat[(size_t)p * 64 + c] = (_Float16)h;

  const float* g2 = gz + (size_t)p * NGATE;
  float cn2 = sigm(g2[32 + c] + 0.01f) * czio[i] + sigm(g2[c]) * tanhx(g2[64 + c]);
  float h2  = sigm(g2[96 + c]) * tanhx(cn2);
  czio[i] = cn2;
  hzio[i] = (_Float16)h2;
  ycat[(size_t)p * 64 + 32 + c] = (_Float16)h2;
}

// ---------------------------------------------------------------------------
// Build NHWC f16 cat = [x, hx, hz, pad] (96 ch).
// Layer 0: x is a 1-channel f32 frame (ch0), pad ch65..95.
// Layers>=1: x is 32-ch f16 y from previous layer.
// ---------------------------------------------------------------------------
__global__ void cat_kernel(const float* __restrict__ xframe, const _Float16* __restrict__ xh,
                           const _Float16* __restrict__ hx, const _Float16* __restrict__ hz,
                           _Float16* __restrict__ cat)
{
  int i = blockIdx.x * blockDim.x + threadIdx.x;
  if (i >= NPIX * CINP0) return;
  const int p = i / CINP0;
  const int c = i - p * CINP0;
  _Float16 v = (_Float16)0.f;
  if (xframe) {
    if (c == 0)       v = (_Float16)xframe[p];
    else if (c <= 32) v = hx[(size_t)p * 32 + (c - 1)];
    else if (c <= 64) v = hz[(size_t)p * 32 + (c - 33)];
  } else {
    if (c < 32)       v = xh[(size_t)p * 32 + c];
    else if (c < 64)  v = hx[(size_t)p * 32 + (c - 32)];
    else              v = hz[(size_t)p * 32 + (c - 64)];
  }
  cat[i] = v;
}

// ---------------------------------------------------------------------------
// 1x1 w_last (32 -> 1): writes feedback frame and/or d_out slot.
// d_out layout: [B, out_len=10, 1, 64, 64].
// ---------------------------------------------------------------------------
__global__ void last_kernel(const _Float16* __restrict__ yin, const float* __restrict__ wl,
                            float* __restrict__ frame_out, float* __restrict__ dout, int jout)
{
  int p = blockIdx.x * blockDim.x + threadIdx.x;
  if (p >= NPIX) return;
  float s = 0.f;
#pragma unroll
  for (int c = 0; c < 32; ++c) s += wl[c] * (float)yin[(size_t)p * 32 + c];
  if (frame_out) frame_out[p] = s;
  if (dout) {
    const int b = p >> 12, yx = p & 4095;
    dout[((size_t)b * 10 + jout) * 4096 + yx] = s;
  }
}

// Copy input_seq [B,T,1,H,W] -> frames[t][b][y][x] (f32).
__global__ void frames_init(const float* __restrict__ in, float* __restrict__ frames)
{
  int i = blockIdx.x * blockDim.x + threadIdx.x;
  if (i >= 10 * NPIX) return;
  const int t = i / NPIX, p = i - t * NPIX;
  const int b = p >> 12, yx = p & 4095;
  frames[(size_t)t * NPIX + p] = in[((size_t)b * 10 + t) * 4096 + yx];
}

// ---------------------------------------------------------------------------
extern "C" void kernel_launch(void* const* d_in, const int* in_sizes, int n_in,
                              void* d_out, int out_size, void* d_ws, size_t ws_size,
                              hipStream_t stream)
{
  (void)in_sizes; (void)n_in; (void)out_size; (void)ws_size;
  const float* input_seq = (const float*)d_in[0];
  const float* w_x0 = (const float*)d_in[1];  const float* b_x0 = (const float*)d_in[2];
  const float* w_z0 = (const float*)d_in[3];  const float* b_z0 = (const float*)d_in[4];
  const float* w_y0 = (const float*)d_in[5];  const float* b_y0 = (const float*)d_in[6];
  const float* w_x1 = (const float*)d_in[7];  const float* b_x1 = (const float*)d_in[8];
  const float* w_z1 = (const float*)d_in[9];  const float* b_z1 = (const float*)d_in[10];
  const float* w_y1 = (const float*)d_in[11]; const float* b_y1 = (const float*)d_in[12];
  const float* w_last = (const float*)d_in[13];
  float* out = (float*)d_out;

  // ---- workspace carve (bump allocator, 256B aligned) ----
  char* ws = (char*)d_ws;
  auto alloc = [&](size_t bytes) -> char* {
    char* p = ws; ws += (bytes + 255) & ~(size_t)255; return p;
  };
  float*    frames = (float*)   alloc((size_t)20 * NPIX * 4);
  _Float16* cat    = (_Float16*)alloc((size_t)NPIX * CINP0 * 2);
  _Float16* ycat   = (_Float16*)alloc((size_t)NPIX * CINY * 2);
  float*    gx     = (float*)   alloc((size_t)NPIX * NGATE * 4);
  float*    gz     = (float*)   alloc((size_t)NPIX * NGATE * 4);

  char* state_base = ws;                              // zero-init region
  float* c_x[3][3]; float* c_z[3];
  _Float16* h_x[3][3]; _Float16* h_z[3];
  for (int l = 0; l < 3; ++l)
    for (int s = 0; s < 3; ++s) c_x[l][s] = (float*)alloc((size_t)NPIX * 32 * 4);
  for (int l = 0; l < 3; ++l) c_z[l] = (float*)alloc((size_t)NPIX * 32 * 4);
  for (int l = 0; l < 3; ++l)
    for (int s = 0; s < 3; ++s) h_x[l][s] = (_Float16*)alloc((size_t)NPIX * 32 * 2);
  for (int l = 0; l < 3; ++l) h_z[l] = (_Float16*)alloc((size_t)NPIX * 32 * 2);
  size_t state_bytes = (size_t)(ws - state_base);

  _Float16* ybuf[2][3];
  for (int p = 0; p < 2; ++p)
    for (int s = 0; s < 3; ++s) ybuf[p][s] = (_Float16*)alloc((size_t)NPIX * 32 * 2);

  // Packed weights: x: KC=27, z: KC=75 (NBT=8); y: KC=2 (NBT=2)
  const int PKX = 27 * 8 * 512, PKZ = 75 * 8 * 512, PKY = 2 * 2 * 512;
  _Float16* wxp[3][3]; _Float16* wzp[3][3]; _Float16* wyp[3][3];
  for (int l = 0; l < 3; ++l)
    for (int s = 0; s < 3; ++s) {
      wxp[l][s] = (_Float16*)alloc((size_t)PKX * 2);
      wzp[l][s] = (_Float16*)alloc((size_t)PKZ * 2);
      wyp[l][s] = (_Float16*)alloc((size_t)PKY * 2);
    }

  hipMemsetAsync(state_base, 0, state_bytes, stream);
  frames_init<<<(10 * NPIX + 255) / 256, 256, 0, stream>>>(input_seq, frames);

  // ---- pack weights ----
  for (int l = 0; l < 3; ++l)
    for (int s = 0; s < 3; ++s) {
      const int cin = (l == 0) ? 65 : 96;
      const float* sx = (l == 0) ? w_x0 + (size_t)s * 128 * 65 * 9
                                 : w_x1 + (size_t)((l - 1) * 3 + s) * 128 * 96 * 9;
      const float* sz = (l == 0) ? w_z0 + (size_t)s * 128 * 65 * 25
                                 : w_z1 + (size_t)((l - 1) * 3 + s) * 128 * 96 * 25;
      const float* sy = (l == 0) ? w_y0 + (size_t)s * 32 * 64
                                 : w_y1 + (size_t)((l - 1) * 3 + s) * 32 * 64;
      pack_weights<<<(PKX + 255) / 256, 256, 0, stream>>>(sx, wxp[l][s], 128, cin, 3, CINP0, PKX);
      pack_weights<<<(PKZ + 255) / 256, 256, 0, stream>>>(sz, wzp[l][s], 128, cin, 5, CINP0, PKZ);
      pack_weights<<<(PKY + 255) / 256, 256, 0, stream>>>(sy, wyp[l][s], 32, 64, 1, CINY, PKY);
    }

  // ---- recurrence: steps idx = 2..19 ----
  const int catN = (NPIX * CINP0 + 255) / 256;
  const int gateN = (NPIX * 32 + 255) / 256;
  const int convGrid = NPIX / 128;   // 128 blocks, 2 image rows each
  for (int idx = 2; idx <= 19; ++idx) {
    for (int l = 0; l < 3; ++l) {
      for (int s = 0; s < 3; ++s) {
        const float* xframe = (l == 0) ? frames + (size_t)(idx - 2 + s) * NPIX : nullptr;
        const _Float16* xh  = (l > 0) ? ybuf[(l - 1) & 1][s] : nullptr;
        const int srcs = (s == 0) ? 0 : s - 1;
        cat_kernel<<<catN, 256, 0, stream>>>(xframe, xh, h_x[l][srcs], h_z[l], cat);

        const float* bx = (l == 0) ? b_x0 + s * 128 : b_x1 + ((l - 1) * 3 + s) * 128;
        const float* bz = (l == 0) ? b_z0 + s * 128 : b_z1 + ((l - 1) * 3 + s) * 128;
        const float* by = (l == 0) ? b_y0 + s * 32  : b_y1 + ((l - 1) * 3 + s) * 32;

        conv_wmma<3, CINP0, 128, false><<<convGrid, 256, 0, stream>>>(cat, wxp[l][s], bx, gx);
        conv_wmma<5, CINP0, 128, false><<<convGrid, 256, 0, stream>>>(cat, wzp[l][s], bz, gz);

        gate_kernel<<<gateN, 256, 0, stream>>>(gx, gz, c_x[l][srcs], c_x[l][s], c_z[l],
                                               h_x[l][s], h_z[l], ycat);

        conv_wmma<1, CINY, 32, true><<<convGrid, 256, 0, stream>>>(ycat, wyp[l][s], by,
                                                                   ybuf[l & 1][s]);
      }
    }
    float* fout = (idx >= 9 && idx <= 18) ? frames + (size_t)(idx + 1) * NPIX : nullptr;
    float* dptr = (idx >= 10) ? out : nullptr;
    last_kernel<<<(NPIX + 255) / 256, 256, 0, stream>>>(ybuf[0][2] /* layer2 -> buf0 */,
                                                        w_last, fout, dptr, idx - 10);
  }
}
